// Attention_58600533786988
// MI455X (gfx1250) — matrix operand order repaired and checked by reading, not compiled, not run on hardware
//
#include <hip/hip_runtime.h>
#include <stdint.h>

typedef __bf16 bf16_t;
typedef __attribute__((ext_vector_type(8)))  __bf16 v8bf;
typedef __attribute__((ext_vector_type(16))) __bf16 v16bf;
typedef __attribute__((ext_vector_type(8)))  float  v8f;
typedef __attribute__((ext_vector_type(4)))  unsigned v4u;
typedef __attribute__((ext_vector_type(8)))  unsigned v8u;

#define DIM    1024
#define NSEQ   2048
#define BATCH  2
#define HEADS  16
#define HDIM   64
#define MROWS  (BATCH * NSEQ)   // 4096

// ---------- helpers ----------

__device__ __forceinline__ bf16_t f2bf(float f) {
  unsigned u = __builtin_bit_cast(unsigned, f);
  unsigned r = (u + 0x7FFFu + ((u >> 16) & 1u)) >> 16;
  unsigned short s = (unsigned short)(r & 0xFFFFu);
  return __builtin_bit_cast(bf16_t, s);
}

__device__ __forceinline__ v16bf frag_cat(v8bf lo, v8bf hi) {
  return __builtin_shufflevector(lo, hi, 0,1,2,3,4,5,6,7,8,9,10,11,12,13,14,15);
}

// A fragment (16x32 bf16, row-major source, rowbase = &tile[rowM * stride]):
// lanes 0-15: K = kb+{0..7,16..23};  lanes 16-31: K = kb+{8..15,24..31}
__device__ __forceinline__ v16bf load_frag_a(const bf16_t* rowbase, int kbase, int half) {
  const bf16_t* p = rowbase + kbase + half * 8;
  v8bf lo = *(const v8bf*)(p);
  v8bf hi = *(const v8bf*)(p + 16);
  return frag_cat(lo, hi);
}

// B fragment (32x16 bf16) from K-transposed tile Bt[n][k] (rowbase = &Bt[colN*stride]):
// lanes 0-15: K = kb+{0..7,16..23};  lanes 16-31: K = kb+{8..15,24..31}
__device__ __forceinline__ v16bf load_frag_b(const bf16_t* rowbase, int kbase, int half) {
  const bf16_t* p = rowbase + kbase + half * 8;
  v8bf lo = *(const v8bf*)(p);
  v8bf hi = *(const v8bf*)(p + 16);
  return frag_cat(lo, hi);
}

__device__ __forceinline__ v8f wmma_bf16(v16bf a, v16bf b, v8f c) {
  return __builtin_amdgcn_wmma_f32_16x16x32_bf16(false, a, false, b, (short)0, c, false, false);
}

// xor-butterfly across each 16-lane half (matches C-fragment row striping)
__device__ __forceinline__ float red16_max(float v) {
  v = fmaxf(v, __builtin_bit_cast(float, __builtin_amdgcn_ds_swizzle(__builtin_bit_cast(int, v), 0x041F)));
  v = fmaxf(v, __builtin_bit_cast(float, __builtin_amdgcn_ds_swizzle(__builtin_bit_cast(int, v), 0x081F)));
  v = fmaxf(v, __builtin_bit_cast(float, __builtin_amdgcn_ds_swizzle(__builtin_bit_cast(int, v), 0x101F)));
  v = fmaxf(v, __builtin_bit_cast(float, __builtin_amdgcn_ds_swizzle(__builtin_bit_cast(int, v), 0x201F)));
  return v;
}
__device__ __forceinline__ float red16_sum(float v) {
  v += __builtin_bit_cast(float, __builtin_amdgcn_ds_swizzle(__builtin_bit_cast(int, v), 0x041F));
  v += __builtin_bit_cast(float, __builtin_amdgcn_ds_swizzle(__builtin_bit_cast(int, v), 0x081F));
  v += __builtin_bit_cast(float, __builtin_amdgcn_ds_swizzle(__builtin_bit_cast(int, v), 0x101F));
  v += __builtin_bit_cast(float, __builtin_amdgcn_ds_swizzle(__builtin_bit_cast(int, v), 0x201F));
  return v;
}

// LDS byte offset of a __shared__ object: generic shared address carries the
// LDS offset in bits [31:0] (flat aperture mapping).
__device__ __forceinline__ unsigned lds_off(const void* p) {
  return (unsigned)(uintptr_t)p;
}

// ---------- Tensor Data Mover: 2-D tile load, bf16 elements ----------
// D# groups 0+1 only (<=2D), VADDR2/3 = NULL. One DMA per call; tracked by TENSORcnt.
__device__ __forceinline__ void tdm_load_2d(unsigned ldsByteOff, const bf16_t* gptr,
                                            unsigned tensor_d0, unsigned tensor_d1,
                                            unsigned tile_d0, unsigned tile_d1,
                                            unsigned stride_d0) {
  unsigned long long ga = (unsigned long long)(uintptr_t)gptr;
  v4u g0;
  g0[0] = 1u;                                           // count=1, user desc
  g0[1] = ldsByteOff;                                   // lds_addr
  g0[2] = (unsigned)(ga & 0xFFFFFFFFu);                 // global_addr[31:0]
  g0[3] = (unsigned)((ga >> 32) & 0x1FFFFFFu) | 0x80000000u; // addr[56:32] | type=2
  v8u g1;
  g1[0] = (1u << 16);                                   // data_size = 2 bytes
  g1[1] = (tensor_d0 & 0xFFFFu) << 16;                  // tensor_dim0[15:0]
  g1[2] = (tensor_d0 >> 16) | ((tensor_d1 & 0xFFFFu) << 16);
  g1[3] = (tensor_d1 >> 16) | (tile_d0 << 16);          // tile_dim0
  g1[4] = tile_d1;                                      // tile_dim1 (tile_dim2=0)
  g1[5] = stride_d0;                                    // tensor_dim0_stride[31:0]
  g1[6] = 0;
  g1[7] = 0;
  asm volatile("tensor_load_to_lds %0, %1" :: "s"(g0), "s"(g1) : "memory");
}

// ---------- kernel 1: f32 -> bf16 convert ----------

__global__ void cvt_kernel(const float* __restrict__ in, bf16_t* __restrict__ out, int n) {
  for (int i = blockIdx.x * blockDim.x + threadIdx.x; i < n; i += gridDim.x * blockDim.x)
    out[i] = f2bf(in[i]);
}

// ---------- kernel 2: QKV GEMM  [4096,1024] x [3072,1024]^T ----------
// 128x128 tile / block, BK=32, TDM double-buffered staging.
// Epilogue scatters Q (pre-scaled 1/sqrt(D)), K, V^T as bf16.

__global__ __launch_bounds__(256)
void qkv_gemm_kernel(const bf16_t* __restrict__ X, const bf16_t* __restrict__ W,
                     bf16_t* __restrict__ Q, bf16_t* __restrict__ K, bf16_t* __restrict__ VT) {
  __shared__ bf16_t As[2][128 * 32];
  __shared__ bf16_t Bs[2][128 * 32];
  const int tid  = threadIdx.x;
  const int wave = tid >> 5, lane = tid & 31;
  const int half = lane >> 4, l16 = lane & 15;
  const int waveM = wave >> 2, waveN = wave & 3;
  const int rowBase = blockIdx.x * 128;
  const int colBase = blockIdx.y * 128;
  const bf16_t* Abase = X + (size_t)rowBase * DIM;
  const bf16_t* Bbase = W + (size_t)colBase * DIM;

  v8f acc[4][2];
#pragma unroll
  for (int mi = 0; mi < 4; ++mi)
#pragma unroll
    for (int ni = 0; ni < 2; ++ni)
      acc[mi][ni] = (v8f){0.f,0.f,0.f,0.f,0.f,0.f,0.f,0.f};

  const int KSTEPS = DIM / 32;
  if (wave == 0) {
    tdm_load_2d(lds_off(&As[0][0]), Abase, DIM, MROWS, 32, 128, DIM);
    tdm_load_2d(lds_off(&Bs[0][0]), Bbase, DIM, 3 * DIM, 32, 128, DIM);
  }
  for (int ki = 0; ki < KSTEPS; ++ki) {
    const int buf = ki & 1;
    if (wave == 0) {
      if (ki + 1 < KSTEPS) {
        int k0n = (ki + 1) * 32;
        tdm_load_2d(lds_off(&As[buf ^ 1][0]), Abase + k0n, DIM, MROWS, 32, 128, DIM);
        tdm_load_2d(lds_off(&Bs[buf ^ 1][0]), Bbase + k0n, DIM, 3 * DIM, 32, 128, DIM);
        __builtin_amdgcn_s_wait_tensorcnt(2);   // current tile landed
      } else {
        __builtin_amdgcn_s_wait_tensorcnt(0);
      }
    }
    __syncthreads();
    v16bf b0 = load_frag_b(&Bs[buf][(waveN * 32 +  0 + l16) * 32], 0, half);
    v16bf b1 = load_frag_b(&Bs[buf][(waveN * 32 + 16 + l16) * 32], 0, half);
#pragma unroll
    for (int mi = 0; mi < 4; ++mi) {
      v16bf a = load_frag_a(&As[buf][(waveM * 64 + mi * 16 + l16) * 32], 0, half);
      acc[mi][0] = wmma_bf16(a, b0, acc[mi][0]);
      acc[mi][1] = wmma_bf16(a, b1, acc[mi][1]);
    }
    __syncthreads();   // buffer reuse fence
  }

#pragma unroll
  for (int mi = 0; mi < 4; ++mi)
#pragma unroll
    for (int ni = 0; ni < 2; ++ni)
#pragma unroll
      for (int vr = 0; vr < 8; ++vr) {
        int gr = rowBase + waveM * 64 + mi * 16 + half * 8 + vr;  // token 0..4095
        int gc = colBase + waveN * 32 + ni * 16 + l16;            // feature 0..3071
        float val = acc[mi][ni][vr];
        int b = gr >> 11, n = gr & (NSEQ - 1);
        int h = (gc & (DIM - 1)) >> 6, d = gc & (HDIM - 1);
        size_t bh = (size_t)(b * HEADS + h);
        if (gc < DIM) {
          Q[(bh * NSEQ + n) * HDIM + d] = f2bf(val * 0.125f);     // 1/sqrt(64)
        } else if (gc < 2 * DIM) {
          K[(bh * NSEQ + n) * HDIM + d] = f2bf(val);
        } else {
          VT[bh * (size_t)(NSEQ * HDIM) + (size_t)d * NSEQ + n] = f2bf(val);
        }
      }
}

// ---------- kernel 3: flash attention ----------
// 4 waves; each wave owns 16 q-rows, full D=64. KV tiles of 64, TDM double-buffered.

__global__ __launch_bounds__(128)
void flash_attn_kernel(const bf16_t* __restrict__ Q, const bf16_t* __restrict__ K,
                       const bf16_t* __restrict__ VT, bf16_t* __restrict__ AO) {
  __shared__ bf16_t Ks[2][64 * 64];    // K tile  [m][d]
  __shared__ bf16_t Vs[2][64 * 64];    // V^T tile [d][m]
  __shared__ bf16_t Ps[4][16 * 64];    // per-wave P tile
  const int tid  = threadIdx.x;
  const int wave = tid >> 5, lane = tid & 31;
  const int half = lane >> 4, l16 = lane & 15;
  const int bh    = blockIdx.y;
  const int qBase = blockIdx.x * 64;

  const bf16_t* qp = Q  + (size_t)bh * NSEQ * HDIM;
  const bf16_t* kp = K  + (size_t)bh * NSEQ * HDIM;
  const bf16_t* vp = VT + (size_t)bh * NSEQ * HDIM;   // [64][2048]

  const int qrow = qBase + wave * 16 + l16;
  v16bf qa0 = load_frag_a(qp + (size_t)qrow * HDIM,  0, half);
  v16bf qa1 = load_frag_a(qp + (size_t)qrow * HDIM, 32, half);

  v8f O[4];
  float mrow[8], lrow[8];
#pragma unroll
  for (int jd = 0; jd < 4; ++jd) O[jd] = (v8f){0.f,0.f,0.f,0.f,0.f,0.f,0.f,0.f};
#pragma unroll
  for (int vr = 0; vr < 8; ++vr) { mrow[vr] = -3.0e38f; lrow[vr] = 0.f; }

  const int ITERS = NSEQ / 64;
  if (wave == 0) {
    tdm_load_2d(lds_off(&Ks[0][0]), kp, HDIM, NSEQ, 64, 64, HDIM);
    tdm_load_2d(lds_off(&Vs[0][0]), vp, NSEQ, HDIM, 64, 64, NSEQ);
  }
  for (int it = 0; it < ITERS; ++it) {
    const int buf = it & 1;
    if (wave == 0) {
      if (it + 1 < ITERS) {
        int kvn = (it + 1) * 64;
        tdm_load_2d(lds_off(&Ks[buf ^ 1][0]), kp + (size_t)kvn * HDIM, HDIM, NSEQ, 64, 64, HDIM);
        tdm_load_2d(lds_off(&Vs[buf ^ 1][0]), vp + kvn, NSEQ, HDIM, 64, 64, NSEQ);
        __builtin_amdgcn_s_wait_tensorcnt(2);
      } else {
        __builtin_amdgcn_s_wait_tensorcnt(0);
      }
    }
    __syncthreads();

    // S = (Q/sqrt(D)) K^T  -- 16x64 per wave
    v8f S[4];
#pragma unroll
    for (int jm = 0; jm < 4; ++jm) {
      S[jm] = (v8f){0.f,0.f,0.f,0.f,0.f,0.f,0.f,0.f};
      const bf16_t* krow = &Ks[buf][(jm * 16 + l16) * 64];
      S[jm] = wmma_bf16(qa0, load_frag_b(krow,  0, half), S[jm]);
      S[jm] = wmma_bf16(qa1, load_frag_b(krow, 32, half), S[jm]);
    }

    // online softmax (rows striped: lanes<16 rows 0-7, lanes>=16 rows 8-15)
    float alpha[8];
#pragma unroll
    for (int vr = 0; vr < 8; ++vr) {
      float t = fmaxf(fmaxf(S[0][vr], S[1][vr]), fmaxf(S[2][vr], S[3][vr]));
      t = red16_max(t);
      float mnew = fmaxf(mrow[vr], t);
      alpha[vr] = __expf(mrow[vr] - mnew);
      mrow[vr] = mnew;
      float ps = 0.f;
#pragma unroll
      for (int jm = 0; jm < 4; ++jm) {
        float p = __expf(S[jm][vr] - mnew);
        S[jm][vr] = p;
        ps += p;
      }
      ps = red16_sum(ps);
      lrow[vr] = lrow[vr] * alpha[vr] + ps;
    }

    // P -> LDS (C-layout -> A-layout transpose), rescale O
#pragma unroll
    for (int jm = 0; jm < 4; ++jm)
#pragma unroll
      for (int vr = 0; vr < 8; ++vr)
        Ps[wave][(half * 8 + vr) * 64 + jm * 16 + l16] = f2bf(S[jm][vr]);
#pragma unroll
    for (int jd = 0; jd < 4; ++jd)
#pragma unroll
      for (int vr = 0; vr < 8; ++vr)
        O[jd][vr] *= alpha[vr];
    __syncthreads();

    // O += P @ V
#pragma unroll
    for (int s = 0; s < 2; ++s) {
      v16bf pa = load_frag_a(&Ps[wave][l16 * 64], s * 32, half);
#pragma unroll
      for (int jd = 0; jd < 4; ++jd) {
        v16bf vb = load_frag_b(&Vs[buf][(jd * 16 + l16) * 64], s * 32, half);
        O[jd] = wmma_bf16(pa, vb, O[jd]);
      }
    }
    __syncthreads();   // buffer reuse fence
  }

  const int b = bh >> 4, h = bh & (HEADS - 1);
#pragma unroll
  for (int jd = 0; jd < 4; ++jd)
#pragma unroll
    for (int vr = 0; vr < 8; ++vr) {
      int n = qBase + wave * 16 + half * 8 + vr;
      int c = h * HDIM + jd * 16 + l16;
      AO[((size_t)b * NSEQ + n) * DIM + c] = f2bf(O[jd][vr] / lrow[vr]);
    }
}

// ---------- kernel 4: projection GEMM + bias ----------

__global__ __launch_bounds__(256)
void proj_gemm_kernel(const bf16_t* __restrict__ X, const bf16_t* __restrict__ W,
                      const float* __restrict__ bias, float* __restrict__ out) {
  __shared__ bf16_t As[2][128 * 32];
  __shared__ bf16_t Bs[2][128 * 32];
  const int tid  = threadIdx.x;
  const int wave = tid >> 5, lane = tid & 31;
  const int half = lane >> 4, l16 = lane & 15;
  const int waveM = wave >> 2, waveN = wave & 3;
  const int rowBase = blockIdx.x * 128;
  const int colBase = blockIdx.y * 128;
  const bf16_t* Abase = X + (size_t)rowBase * DIM;
  const bf16_t* Bbase = W + (size_t)colBase * DIM;

  v8f acc[4][2];
#pragma unroll
  for (int mi = 0; mi < 4; ++mi)
#pragma unroll
    for (int ni = 0; ni < 2; ++ni)
      acc[mi][ni] = (v8f){0.f,0.f,0.f,0.f,0.f,0.f,0.f,0.f};

  const int KSTEPS = DIM / 32;
  if (wave == 0) {
    tdm_load_2d(lds_off(&As[0][0]), Abase, DIM, MROWS, 32, 128, DIM);
    tdm_load_2d(lds_off(&Bs[0][0]), Bbase, DIM, DIM, 32, 128, DIM);
  }
  for (int ki = 0; ki < KSTEPS; ++ki) {
    const int buf = ki & 1;
    if (wave == 0) {
      if (ki + 1 < KSTEPS) {
        int k0n = (ki + 1) * 32;
        tdm_load_2d(lds_off(&As[buf ^ 1][0]), Abase + k0n, DIM, MROWS, 32, 128, DIM);
        tdm_load_2d(lds_off(&Bs[buf ^ 1][0]), Bbase + k0n, DIM, DIM, 32, 128, DIM);
        __builtin_amdgcn_s_wait_tensorcnt(2);
      } else {
        __builtin_amdgcn_s_wait_tensorcnt(0);
      }
    }
    __syncthreads();
    v16bf b0 = load_frag_b(&Bs[buf][(waveN * 32 +  0 + l16) * 32], 0, half);
    v16bf b1 = load_frag_b(&Bs[buf][(waveN * 32 + 16 + l16) * 32], 0, half);
#pragma unroll
    for (int mi = 0; mi < 4; ++mi) {
      v16bf a = load_frag_a(&As[buf][(waveM * 64 + mi * 16 + l16) * 32], 0, half);
      acc[mi][0] = wmma_bf16(a, b0, acc[mi][0]);
      acc[mi][1] = wmma_bf16(a, b1, acc[mi][1]);
    }
    __syncthreads();
  }

#pragma unroll
  for (int mi = 0; mi < 4; ++mi)
#pragma unroll
    for (int ni = 0; ni < 2; ++ni)
#pragma unroll
      for (int vr = 0; vr < 8; ++vr) {
        int gr = rowBase + waveM * 64 + mi * 16 + half * 8 + vr;
        int gc = colBase + waveN * 32 + ni * 16 + l16;
        out[(size_t)gr * DIM + gc] = acc[mi][ni][vr] + bias[gc];
      }
}

// ---------- launch ----------

extern "C" void kernel_launch(void* const* d_in, const int* in_sizes, int n_in,
                              void* d_out, int out_size, void* d_ws, size_t ws_size,
                              hipStream_t stream) {
  const float* x      = (const float*)d_in[0];
  const float* w_qkv  = (const float*)d_in[1];
  const float* w_proj = (const float*)d_in[2];
  const float* b_proj = (const float*)d_in[3];
  float* out = (float*)d_out;

  bf16_t* ws     = (bf16_t*)d_ws;
  bf16_t* xb     = ws;                                   // 4096*1024
  bf16_t* wqkvb  = xb     + (size_t)MROWS * DIM;         // 3072*1024
  bf16_t* wprojb = wqkvb  + (size_t)3 * DIM * DIM;       // 1024*1024
  bf16_t* qb     = wprojb + (size_t)DIM * DIM;           // 32*2048*64
  bf16_t* kb     = qb     + (size_t)BATCH * HEADS * NSEQ * HDIM;
  bf16_t* vtb    = kb     + (size_t)BATCH * HEADS * NSEQ * HDIM;
  bf16_t* aob    = vtb    + (size_t)BATCH * HEADS * NSEQ * HDIM;  // 4096*1024
  // total ws use: 48 MB of bf16

  cvt_kernel<<<2048, 256, 0, stream>>>(x,      xb,     MROWS * DIM);
  cvt_kernel<<<1536, 256, 0, stream>>>(w_qkv,  wqkvb,  3 * DIM * DIM);
  cvt_kernel<<<512,  256, 0, stream>>>(w_proj, wprojb, DIM * DIM);

  qkv_gemm_kernel <<<dim3(MROWS / 128, (3 * DIM) / 128), 256, 0, stream>>>(xb, wqkvb, qb, kb, vtb);
  flash_attn_kernel<<<dim3(NSEQ / 64, BATCH * HEADS),    128, 0, stream>>>(qb, kb, vtb, aob);
  proj_gemm_kernel<<<dim3(MROWS / 128, DIM / 128),       256, 0, stream>>>(aob, wprojb, b_proj, out);
}
